// SingleHeadAttention_29540785062259
// MI455X (gfx1250) — matrix operand order repaired and checked by reading, not compiled, not run on hardware
//
#include <hip/hip_runtime.h>
#include <stdint.h>
#include <math.h>

// ---------------------------------------------------------------------------
// FlashAttention-2 forward for gfx1250 (MI455X), wave32, v_wmma_f32_16x16x32_bf16
//   Pre-pass: K -> bf16 Kb[key][d], V -> bf16 VTb[d][key] (tiled transpose)
//   Main:    S^T = K_tile x Q^T ; O^T += V^T x P^T (P^T built in registers)
//   Staging: global_load_async_to_lds_b128 ; software-pipelined WMMA fragments
// exp2-domain online softmax, fp32 accum, fp32 in/out, split-K + combine.
// ---------------------------------------------------------------------------

typedef __attribute__((ext_vector_type(16))) __bf16 v16bf;
typedef __attribute__((ext_vector_type(2)))  __bf16 v2bf;
typedef __attribute__((ext_vector_type(8)))  float  v8f;
typedef __attribute__((ext_vector_type(4)))  int    v4i;

#define DHEAD   128
#define BN      64            // keys per iteration
#define WAVES   4
#define THREADS (WAVES * 32)
#define BM      (WAVES * 16)  // 64 queries per workgroup (16 per wave)
#define KSTR    136           // bf16 row stride (16B-aligned rows, bank-clean)
#define VSTR    72            // bf16 row stride for V^T tile
#define LOG2E   1.4426950408889634f
#define MAXSPLIT 4

#define AS1 __attribute__((address_space(1)))
#define AS3 __attribute__((address_space(3)))

#if defined(__has_builtin)
#  if __has_builtin(__builtin_amdgcn_global_load_async_to_lds_b128)
#    define HAVE_ASYNC_LDS 1
#  endif
#endif

union Frag { uint32_t u[8]; v16bf v; };

__device__ __forceinline__ float fast_exp2(float x) {
#if __has_builtin(__builtin_amdgcn_exp2f)
  return __builtin_amdgcn_exp2f(x);
#else
  return exp2f(x);
#endif
}

__device__ __forceinline__ uint32_t pack_bf16(float lo, float hi) {
#if __has_builtin(__builtin_amdgcn_cvt_pk_bf16_f32)
  union { v2bf v; uint32_t u; } c;
  c.v = __builtin_amdgcn_cvt_pk_bf16_f32(lo, hi);
  return c.u;
#else
  uint32_t a = __float_as_uint(lo), b = __float_as_uint(hi);
  a = (a + 0x7FFFu + ((a >> 16) & 1u)) >> 16;   // RNE
  b = (b + 0x7FFFu + ((b >> 16) & 1u)) >> 16;
  return (a & 0xFFFFu) | (b << 16);
#endif
}

// 16-byte global -> LDS copy: async path if available, sync fallback.
__device__ __forceinline__ void copy16_to_lds(uint16_t* lds, const uint16_t* g) {
#ifdef HAVE_ASYNC_LDS
  __builtin_amdgcn_global_load_async_to_lds_b128((AS1 v4i*)g, (AS3 v4i*)lds, 0, 0);
#else
  *(uint4*)lds = *(const uint4*)g;
#endif
}
__device__ __forceinline__ void wait_lds_copies() {
#ifdef HAVE_ASYNC_LDS
#  if __has_builtin(__builtin_amdgcn_s_wait_asynccnt)
  __builtin_amdgcn_s_wait_asynccnt(0);
#  else
  asm volatile("s_wait_asynccnt 0" ::: "memory");
#  endif
#endif
}

__device__ __forceinline__ void load_frag(Frag& f, const uint16_t* p) {
  *(uint4*)&f.u[0] = *(const uint4*)(p);
  *(uint4*)&f.u[4] = *(const uint4*)(p + 8);
}

// ---------------- pre-pass: K fp32 -> bf16 row-major ----------------
__global__ __launch_bounds__(256)
void conv_k(const float* __restrict__ K, uint16_t* __restrict__ Kb, int n2) {
  int i = blockIdx.x * 256 + threadIdx.x;
  if (i < n2) {
    float2 x = ((const float2*)K)[i];
    ((uint32_t*)Kb)[i] = pack_bf16(x.x, x.y);
  }
}

// ---------------- pre-pass: V fp32 -> bf16 transposed [d][key] ----------------
__global__ __launch_bounds__(256)
void conv_vt(const float* __restrict__ V, uint16_t* __restrict__ VTb, int sk) {
  __shared__ alignas(16) uint16_t tile[64][72];   // [key][d], padded
  const int kb  = blockIdx.x;
  const int db  = blockIdx.y;
  const int tid = threadIdx.x;
  for (int i = tid; i < 64 * 32; i += 256) {      // coalesced fp32 reads
    int r = i >> 5;
    int c = (i & 31) << 1;
    float2 x = *(const float2*)(V + ((size_t)(kb * 64 + r)) * DHEAD + db * 64 + c);
    *(uint32_t*)&tile[r][c] = pack_bf16(x.x, x.y);
  }
  __syncthreads();
  for (int i = tid; i < 64 * 32; i += 256) {      // coalesced bf16-pair writes
    int d  = i >> 5;
    int k2 = (i & 31) << 1;
    uint32_t pk = (uint32_t)tile[k2][d] | ((uint32_t)tile[k2 + 1][d] << 16);
    *(uint32_t*)&VTb[(size_t)(db * 64 + d) * sk + kb * 64 + k2] = pk;
  }
}

// ---------------- main flash-attention kernel ----------------
template <int PRECONV>
__global__ __launch_bounds__(THREADS)
void fa_fwd(const float* __restrict__ Q, const float* __restrict__ K,
            const float* __restrict__ V, float* __restrict__ O,
            float* __restrict__ Opart, float* __restrict__ Mst,
            float* __restrict__ Lst,
            const uint16_t* __restrict__ Kb, const uint16_t* __restrict__ VTb,
            int nkb_local, int sq, int sk)
{
  __shared__ alignas(16) uint16_t sK [BN    * KSTR];  // K tile, row-major bf16
  __shared__ alignas(16) uint16_t sVT[DHEAD * VSTR];  // V tile, transposed bf16

  const int tid  = threadIdx.x;
  const int wave = tid >> 5;
  const int lane = tid & 31;
  const int lq   = lane & 15;   // this lane's query column (C/D layout N)
  const int h    = lane >> 4;   // lane half

  const int qrow = blockIdx.x * BM + wave * 16 + lq;
  const float* qp = Q + (size_t)qrow * DHEAD;

  // --- Q^T B-fragments, pre-scaled by log2(e). B layout: reg w = d pair 2w+16h ---
  Frag qf[4];
#pragma unroll
  for (int dd = 0; dd < 4; ++dd) {
    const float4* qv = (const float4*)(qp + 16 * h + 32 * dd);
#pragma unroll
    for (int w2 = 0; w2 < 4; ++w2) {
      float4 x = qv[w2];
      qf[dd].u[2 * w2]     = pack_bf16(x.x * LOG2E, x.y * LOG2E);
      qf[dd].u[2 * w2 + 1] = pack_bf16(x.z * LOG2E, x.w * LOG2E);
    }
  }

  v8f oacc[8];                       // O^T accumulators: rows d=v+8h+16j, col q=lq
#pragma unroll
  for (int j = 0; j < 8; ++j)
#pragma unroll
    for (int v = 0; v < 8; ++v) oacc[j][v] = 0.f;

  float mrow = -__builtin_inff();
  float lrow = 0.f;

  const int kb0   = blockIdx.y * nkb_local;
  const int kbend = kb0 + nkb_local;

  const uint16_t* kfrag_base = &sK [lq * KSTR + 16 * h];
  const uint16_t* vfrag_base = &sVT[lq * VSTR + 16 * h];

  for (int kb = kb0; kb < kbend; ++kb) {
    if (PRECONV) {
      // ---- pure b128 async copies from pre-converted bf16 buffers ----
      const uint16_t* kgb = Kb + (size_t)kb * BN * DHEAD;
#pragma unroll
      for (int i = tid; i < BN * (DHEAD / 8); i += THREADS) {   // 8 per thread
        int row = i >> 4;
        int c   = (i & 15) << 3;
        copy16_to_lds(&sK[row * KSTR + c], kgb + row * DHEAD + c);
      }
      const uint16_t* vgb = VTb + (size_t)kb * BN;
#pragma unroll
      for (int i = tid; i < DHEAD * (BN / 8); i += THREADS) {   // 8 per thread
        int d = i >> 3;
        int c = (i & 7) << 3;
        copy16_to_lds(&sVT[d * VSTR + c], vgb + (size_t)d * sk + c);
      }
      if (kb + 1 < kbend) {
        __builtin_prefetch(Kb + (size_t)(kb + 1) * BN * DHEAD + tid * 64, 0, 1);
        __builtin_prefetch(VTb + (size_t)tid * sk + (kb + 1) * BN, 0, 1);
      }
      wait_lds_copies();
    } else {
      // ---- fallback: convert fp32 -> bf16 while staging ----
      const float* kg = K + (size_t)kb * BN * DHEAD;
#pragma unroll 4
      for (int i = tid; i < BN * (DHEAD / 4); i += THREADS) {
        int row = i >> 5;
        int p   = (i & 31) << 2;
        float4 x = *(const float4*)(kg + row * DHEAD + p);
        uint2 pk;
        pk.x = pack_bf16(x.x, x.y);
        pk.y = pack_bf16(x.z, x.w);
        *(uint2*)&sK[row * KSTR + p] = pk;
      }
      const float* vg = V + (size_t)kb * BN * DHEAD;
#pragma unroll 4
      for (int i = tid; i < (BN / 2) * DHEAD; i += THREADS) {
        int row = (i / DHEAD) * 2;
        int d   = i % DHEAD;
        float a = vg[row * DHEAD + d];
        float b = vg[(row + 1) * DHEAD + d];
        *(uint32_t*)&sVT[d * VSTR + row] = pack_bf16(a, b);
      }
      if (kb + 1 < kbend) {
        __builtin_prefetch(K + (size_t)(kb + 1) * BN * DHEAD + tid * 64, 0, 1);
        __builtin_prefetch(V + (size_t)(kb + 1) * BN * DHEAD + tid * 64, 0, 1);
      }
    }
    __syncthreads();

    // ---- S^T: 16 WMMAs, ping-pong fragment pipeline, t interleaved ----
    v8f st[4];
#pragma unroll
    for (int t = 0; t < 4; ++t)
#pragma unroll
      for (int v = 0; v < 8; ++v) st[t][v] = 0.f;

    Frag kf[2];
    // step s: t = s&3 (key tile), dd = s>>2 (K-step over D)
    load_frag(kf[0], kfrag_base);
#pragma unroll
    for (int s = 0; s < 16; ++s) {
      if (s + 1 < 16) {
        int t1 = (s + 1) & 3, dd1 = (s + 1) >> 2;
        load_frag(kf[(s + 1) & 1], kfrag_base + (16 * t1) * KSTR + 32 * dd1);
      }
      int t = s & 3, dd = s >> 2;
      st[t] = __builtin_amdgcn_wmma_f32_16x16x32_bf16(
          false, kf[s & 1].v, false, qf[dd].v, (short)0, st[t], false, false);
    }

    // ---- exp2-domain online softmax ----
    float bmax = st[0][0];
#pragma unroll
    for (int t = 0; t < 4; ++t)
#pragma unroll
      for (int v = 0; v < 8; ++v) bmax = fmaxf(bmax, st[t][v]);
    bmax = fmaxf(bmax, __shfl_xor(bmax, 16, 32));
    float mnew  = fmaxf(mrow, bmax);
    float alpha = fast_exp2(mrow - mnew);
    mrow = mnew;

    float bsum = 0.f;
#pragma unroll
    for (int t = 0; t < 4; ++t)
#pragma unroll
      for (int v = 0; v < 8; ++v) {
        float p = fast_exp2(st[t][v] - mnew);
        st[t][v] = p;
        bsum += p;
      }
    bsum += __shfl_xor(bsum, 16, 32);
    lrow = lrow * alpha + bsum;
#pragma unroll
    for (int j = 0; j < 8; ++j)
#pragma unroll
      for (int v = 0; v < 8; ++v) oacc[j][v] *= alpha;

    // ---- build both P^T B-frags up-front (registers + one half-swap) ----
    Frag pf[2];
#pragma unroll
    for (int kk = 0; kk < 2; ++kk) {
      const int t_own = 2 * kk + h;
      const int t_oth = 2 * kk + (1 - h);
      uint32_t own_pk[4], src_pk[4], e_pk[4];
#pragma unroll
      for (int w = 0; w < 4; ++w) {
        own_pk[w] = pack_bf16(st[t_own][2 * w], st[t_own][2 * w + 1]);
        src_pk[w] = pack_bf16(st[t_oth][2 * w], st[t_oth][2 * w + 1]);
      }
#pragma unroll
      for (int w = 0; w < 4; ++w)
        e_pk[w] = (uint32_t)__shfl_xor((int)src_pk[w], 16, 32);
#pragma unroll
      for (int w = 0; w < 4; ++w) {
        pf[kk].u[w]     = h ? e_pk[w]   : own_pk[w];
        pf[kk].u[4 + w] = h ? own_pk[w] : e_pk[w];
      }
    }

    // ---- O^T += V^T x P^T: 16 WMMAs, ping-pong fragment pipeline ----
    Frag vf[2];
    // step s: kk = s>>3 (key half), j = s&7 (d tile)
    load_frag(vf[0], vfrag_base);
#pragma unroll
    for (int s = 0; s < 16; ++s) {
      if (s + 1 < 16) {
        int kk1 = (s + 1) >> 3, j1 = (s + 1) & 7;
        load_frag(vf[(s + 1) & 1], vfrag_base + (16 * j1) * VSTR + 32 * kk1);
      }
      int kk = s >> 3, j = s & 7;
      oacc[j] = __builtin_amdgcn_wmma_f32_16x16x32_bf16(
          false, vf[s & 1].v, false, pf[kk].v, (short)0, oacc[j], false, false);
    }
    __syncthreads();
  }

  // ---- epilogue ----
  if (gridDim.y == 1) {
    float invl = 1.0f / lrow;
    float* op = O + (size_t)qrow * DHEAD;
#pragma unroll
    for (int j = 0; j < 8; ++j) {
      float4 a, b;
      a.x = oacc[j][0] * invl; a.y = oacc[j][1] * invl;
      a.z = oacc[j][2] * invl; a.w = oacc[j][3] * invl;
      b.x = oacc[j][4] * invl; b.y = oacc[j][5] * invl;
      b.z = oacc[j][6] * invl; b.w = oacc[j][7] * invl;
      *(float4*)(op + 16 * j + 8 * h)     = a;
      *(float4*)(op + 16 * j + 8 * h + 4) = b;
    }
  } else {
    const size_t sbase = (size_t)blockIdx.y * sq + qrow;
    float* op = Opart + sbase * DHEAD;
#pragma unroll
    for (int j = 0; j < 8; ++j) {
      float4 a, b;
      a.x = oacc[j][0]; a.y = oacc[j][1]; a.z = oacc[j][2]; a.w = oacc[j][3];
      b.x = oacc[j][4]; b.y = oacc[j][5]; b.z = oacc[j][6]; b.w = oacc[j][7];
      *(float4*)(op + 16 * j + 8 * h)     = a;
      *(float4*)(op + 16 * j + 8 * h + 4) = b;
    }
    if (lane < 16) {
      Mst[sbase] = mrow;
      Lst[sbase] = lrow;
    }
  }
}

// Combine split-K partials: O = sum_s O~_s*2^(m_s-M) / sum_s l_s*2^(m_s-M)
__global__ __launch_bounds__(256)
void fa_combine(const float* __restrict__ Opart, const float* __restrict__ Mst,
                const float* __restrict__ Lst, float* __restrict__ O,
                int splits, int sq)
{
  int idx = blockIdx.x * 256 + threadIdx.x;
  if (idx >= sq * DHEAD) return;
  int q = idx / DHEAD;
  float M = -__builtin_inff();
  for (int s = 0; s < splits; ++s) M = fmaxf(M, Mst[(size_t)s * sq + q]);
  float L = 0.f, acc = 0.f;
  for (int s = 0; s < splits; ++s) {
    float w = fast_exp2(Mst[(size_t)s * sq + q] - M);
    L   += Lst[(size_t)s * sq + q] * w;
    acc += Opart[((size_t)s * sq + q) * DHEAD + (idx % DHEAD)] * w;
  }
  O[idx] = acc / L;
}

extern "C" void kernel_launch(void* const* d_in, const int* in_sizes, int n_in,
                              void* d_out, int out_size, void* d_ws, size_t ws_size,
                              hipStream_t stream) {
  const float* Q = (const float*)d_in[0];
  const float* K = (const float*)d_in[1];
  const float* V = (const float*)d_in[2];
  float* O = (float*)d_out;
  const int sq  = in_sizes[0] / DHEAD;   // 8192
  const int sk  = in_sizes[1] / DHEAD;   // 8192
  const int nkb = sk / BN;               // 128 key blocks

  // Workspace layout: [Kb bf16][VTb bf16][Opart f32][Mst f32][Lst f32]
  const size_t conv_bytes = (size_t)sk * DHEAD * 2 * sizeof(uint16_t);
  const int preconv = (ws_size >= conv_bytes) ? 1 : 0;
  uint16_t* Kb  = (uint16_t*)d_ws;
  uint16_t* VTb = Kb + (size_t)sk * DHEAD;

  const size_t part_bytes = (size_t)MAXSPLIT * sq * DHEAD * sizeof(float)
                          + (size_t)MAXSPLIT * sq * 2 * sizeof(float);
  const size_t part_off = preconv ? conv_bytes : 0;
  int splits = 1;
  if (ws_size >= part_off + part_bytes && (nkb % MAXSPLIT) == 0) splits = MAXSPLIT;

  float* Opart = (float*)((char*)d_ws + part_off);
  float* Mst   = Opart + (size_t)splits * sq * DHEAD;
  float* Lst   = Mst   + (size_t)splits * sq;

  dim3 grid(sq / BM, splits), block(THREADS);
  if (preconv) {
    int n2 = sk * DHEAD / 2;
    conv_k<<<(n2 + 255) / 256, 256, 0, stream>>>(K, Kb, n2);
    conv_vt<<<dim3(sk / 64, DHEAD / 64), 256, 0, stream>>>(V, VTb, sk);
    fa_fwd<1><<<grid, block, 0, stream>>>(Q, K, V, O, Opart, Mst, Lst,
                                          Kb, VTb, nkb / splits, sq, sk);
  } else {
    fa_fwd<0><<<grid, block, 0, stream>>>(Q, K, V, O, Opart, Mst, Lst,
                                          Kb, VTb, nkb / splits, sq, sk);
  }
  if (splits > 1) {
    int n = sq * DHEAD;
    fa_combine<<<(n + 255) / 256, 256, 0, stream>>>(Opart, Mst, Lst, O, splits, sq);
  }
}
